// AutoregressiveBeamDecoder_19679540150603
// MI455X (gfx1250) — matrix operand order repaired and checked by reading, not compiled, not run on hardware
//
#include <hip/hip_runtime.h>
#include <hip/hip_bf16.h>

// ---------------------------------------------------------------------------
// Problem constants (from reference): B=256, T=128, D=512, H=1024, NB=64, HH=8
// ---------------------------------------------------------------------------
#define B_ 256
#define T_ 128
#define D_ 512
#define H_ 1024
#define NB_ 64
#define HH_ 8
#define LN_EPS 1e-5f

typedef __attribute__((ext_vector_type(16))) __bf16 bf16x16;
typedef __attribute__((ext_vector_type(8)))  float  floatx8;
typedef __attribute__((ext_vector_type(4)))  int    v4i;

union FragU {
    uint4   u[2];
    bf16x16 v;
};

__device__ __forceinline__ unsigned short f2bf(float f) {
    unsigned int u = __float_as_uint(f);
    unsigned int r = (u + 0x7FFFu + ((u >> 16) & 1u)) >> 16;   // round-to-nearest-even
    return (unsigned short)r;
}

__device__ __forceinline__ float sigmoidf(float x) {
    return 1.0f / (1.0f + __expf(-x));
}

// ---------------------------------------------------------------------------
// gfx1250 async global->LDS copy (ASYNCcnt path), with synchronous fallback
//   builtin signature (probe-confirmed): (v4i AS1*, v4i AS3*, imm off, imm cpol)
// ---------------------------------------------------------------------------
#if defined(__has_builtin)
#  if __has_builtin(__builtin_amdgcn_global_load_async_to_lds_b128)
#    define HAVE_ASYNC_LDS 1
#  endif
#endif
#ifndef HAVE_ASYNC_LDS
#  define HAVE_ASYNC_LDS 0
#endif

#if HAVE_ASYNC_LDS
typedef __attribute__((address_space(1))) v4i* gv4i_p;
typedef __attribute__((address_space(3))) v4i* lv4i_p;

__device__ __forceinline__ void copy16(const unsigned short* g, unsigned short* l) {
    __builtin_amdgcn_global_load_async_to_lds_b128((gv4i_p)g, (lv4i_p)l, 0, 0);
}
__device__ __forceinline__ void wait_async() {
#  if __has_builtin(__builtin_amdgcn_s_wait_asynccnt)
    __builtin_amdgcn_s_wait_asynccnt(0);
#  else
    asm volatile("s_wait_asynccnt 0" ::: "memory");
#  endif
}
#else
__device__ __forceinline__ void copy16(const unsigned short* g, unsigned short* l) {
    *(uint4*)l = *(const uint4*)g;
}
__device__ __forceinline__ void wait_async() {}
#endif

// ---------------------------------------------------------------------------
// Generic f32 -> bf16 conversion (weights)
// ---------------------------------------------------------------------------
__global__ __launch_bounds__(256) void cvt_bf16_kernel(const float* __restrict__ src,
                                                       unsigned short* __restrict__ dst,
                                                       int n) {
    int i = blockIdx.x * 256 + threadIdx.x;
    if (i < n) dst[i] = f2bf(src[i]);
}

// ---------------------------------------------------------------------------
// Init: ctx_global = mean_t ctx[b,:,d]; hist_emb = mean_j emb[hist[b,j],d]
//       Ain0[b, 0:D) = ctx_global ; Ain0[b, D:2D) = hist_emb   (bf16)
//       prev[b] = hist[b, HH-1]
// ---------------------------------------------------------------------------
__global__ __launch_bounds__(256) void init_kernel(const float* __restrict__ ctx,
                                                   const int* __restrict__ hist,
                                                   const float* __restrict__ emb,
                                                   unsigned short* __restrict__ Ain0,
                                                   int* __restrict__ prev) {
    int i = blockIdx.x * 256 + threadIdx.x;      // over B*D = 131072
    int b = i >> 9;                              // D_ = 512
    int d = i & (D_ - 1);

    const float* p = ctx + (size_t)b * T_ * D_ + d;
    float s = 0.f;
#pragma unroll 4
    for (int t = 0; t < T_; ++t) s += p[(size_t)t * D_];
    float cg = s * (1.0f / T_);

    float e = 0.f;
#pragma unroll
    for (int j = 0; j < HH_; ++j) e += emb[(size_t)hist[b * HH_ + j] * D_ + d];
    float he = e * (1.0f / HH_);

    Ain0[(size_t)b * (2 * D_) + d]      = f2bf(cg);
    Ain0[(size_t)b * (2 * D_) + D_ + d] = f2bf(he);
    if (d == 0) prev[b] = hist[b * HH_ + HH_ - 1];
}

// ---------------------------------------------------------------------------
// Per-step input concat: Ain[b, 0:D) = ctx[b,t,:] ; Ain[b, D:2D) = emb[prev[b],:]
// ---------------------------------------------------------------------------
__global__ __launch_bounds__(256) void concat_in_kernel(const float* __restrict__ ctx,
                                                        const float* __restrict__ emb,
                                                        const int* __restrict__ prev,
                                                        unsigned short* __restrict__ Ain,
                                                        int t) {
    int i = blockIdx.x * 256 + threadIdx.x;      // over B*2D = 262144
    int b = i >> 10;                             // 2D = 1024
    int c = i & 1023;
    float v;
    if (c < D_) v = ctx[(size_t)b * T_ * D_ + (size_t)t * D_ + c];
    else        v = emb[(size_t)prev[b] * D_ + (c - D_)];
    Ain[i] = f2bf(v);
}

// ---------------------------------------------------------------------------
// bf16 WMMA GEMM:  C[M=256 x N] = act( A[M x K] * W[N x K]^T + bias )
//   Requires N % 128 == 0, K % 64 == 0.  A, W bf16 (ushort), K contiguous.
//   Block tile 64x128, K chunked by 64, DOUBLE-BUFFERED async LDS staging:
//   async copies for chunk k+1 are issued before the 8 WMMAs of chunk k.
//   act: 0 = none, 1 = relu, 2 = tanh
// ---------------------------------------------------------------------------
#define TM 64
#define TN 128
#define TK 64
#define ASTRIDE 72   // bf16 elems per LDS row (64 + 8 pad)
#define BSTRIDE 72

__device__ __forceinline__ void stage_tiles(const unsigned short* __restrict__ A, int lda,
                                            const unsigned short* __restrict__ W, int ldw,
                                            int block_m, int block_n, int kk, int K, int tid,
                                            unsigned short* __restrict__ As,
                                            unsigned short* __restrict__ Bs) {
    // A tile: 64 rows x 64 bf16 (2 x 16B per thread)
#pragma unroll
    for (int it = 0; it < 2; ++it) {
        int idx = tid + it * 256;
        int r = idx >> 3;
        int c = (idx & 7) * 8;
        copy16(A + (size_t)(block_m + r) * lda + kk + c, &As[r * ASTRIDE + c]);
    }
    // W tile: 128 rows x 64 bf16 (4 x 16B per thread)
#pragma unroll
    for (int it = 0; it < 4; ++it) {
        int idx = tid + it * 256;
        int r = idx >> 3;
        int c = (idx & 7) * 8;
        copy16(W + (size_t)(block_n + r) * ldw + kk + c, &Bs[r * BSTRIDE + c]);
    }
    // prefetch the chunk after this one (global_prefetch_b8)
    if (kk + TK < K) {
        int n = block_n + (tid >> 1);
        __builtin_prefetch(W + (size_t)n * ldw + kk + TK + (tid & 1) * 32, 0, 1);
    }
}

__global__ __launch_bounds__(256)
void gemm_bf16_wmma(const unsigned short* __restrict__ A, int lda,
                    const unsigned short* __restrict__ W, int ldw,
                    const float* __restrict__ bias,
                    float* __restrict__ Cf, long long ldc,
                    unsigned short* __restrict__ Cbf, int ldcbf,
                    int N, int K, int act) {
    __shared__ __align__(16) unsigned short As[2][TM * ASTRIDE];
    __shared__ __align__(16) unsigned short Bs[2][TN * BSTRIDE];

    const int tid  = threadIdx.x;
    const int lane = tid & 31;
    const int wave = tid >> 5;
    const int wm   = wave & 3;        // row tile within block (0..3)
    const int wn   = wave >> 2;       // 64-col group (0..1)
    const int block_m = blockIdx.y * TM;
    const int block_n = blockIdx.x * TN;

    const int l  = lane & 15;
    const int kh = lane >> 4;

    floatx8 acc[4] = {};

    // prologue: stage chunk 0
    stage_tiles(A, lda, W, ldw, block_m, block_n, 0, K, tid, As[0], Bs[0]);
    wait_async();
    __syncthreads();

    int buf = 0;
    for (int k0 = 0; k0 < K; k0 += TK) {
        const int  nxt      = buf ^ 1;
        const bool has_next = (k0 + TK) < K;

        // issue async staging of the NEXT chunk before computing this one
        if (has_next)
            stage_tiles(A, lda, W, ldw, block_m, block_n, k0 + TK, K, tid, As[nxt], Bs[nxt]);

        // ---- compute current chunk: two K=32 sub-chunks, 4 N-tiles => 8 WMMA ----
        const unsigned short* Ab = As[buf];
        const unsigned short* Bb = Bs[buf];
#pragma unroll
        for (int ks = 0; ks < TK; ks += 32) {
            // A fragment (16x32): lane<16 -> K 0..15; lane>=16 -> K 16..31
            FragU a;
            const unsigned short* pa = &Ab[(wm * 16 + l) * ASTRIDE + ks];
            a.u[0] = *(const uint4*)(pa + kh * 16);
            a.u[1] = *(const uint4*)(pa + kh * 16 + 8);
#pragma unroll
            for (int j = 0; j < 4; ++j) {
                FragU bm;
                const unsigned short* pb = &Bb[(wn * 64 + j * 16 + l) * BSTRIDE + ks + kh * 16];
                bm.u[0] = *(const uint4*)(pb);
                bm.u[1] = *(const uint4*)(pb + 8);
                acc[j] = __builtin_amdgcn_wmma_f32_16x16x32_bf16(
                    false, a.v, false, bm.v, (short)0, acc[j], false, false);
            }
        }

        if (has_next) {
            wait_async();       // next-chunk copies complete (overlapped with WMMA above)
            __syncthreads();
        }
        buf = nxt;
    }

    // ---- epilogue: D layout => lanes 0-15: rows 0..7, lanes 16-31: rows 8..15
    const int row0 = block_m + wm * 16 + kh * 8;
#pragma unroll
    for (int j = 0; j < 4; ++j) {
        int col = block_n + wn * 64 + j * 16 + l;
        float bv = bias ? bias[col] : 0.0f;
#pragma unroll
        for (int r = 0; r < 8; ++r) {
            float v = acc[j][r] + bv;
            if (act == 1)      v = fmaxf(v, 0.0f);
            else if (act == 2) v = tanhf(v);
            int row = row0 + r;
            if (Cf)  Cf[(size_t)row * ldc + col] = v;
            if (Cbf) Cbf[(size_t)row * ldcbf + col] = f2bf(v);
        }
    }
}

// ---------------------------------------------------------------------------
// Fused logits + argmax:  logits[b,t,:] = Obf[b,:] @ W_o2^T + b_o2 (NB=64)
//   followed by per-row argmax -> prev[b].  One block per 64 batch rows;
//   8 waves, each wave 16 rows x 32 cols (2 WMMA accs), K = H chunked by 64.
// ---------------------------------------------------------------------------
__global__ __launch_bounds__(256)
void logits_argmax_kernel(const unsigned short* __restrict__ A,   // Obf, lda = H
                          const unsigned short* __restrict__ W,   // W_o2 bf16, 64 x H
                          const float* __restrict__ bias,
                          float* __restrict__ out, int t,
                          int* __restrict__ prev) {
    __shared__ __align__(16) unsigned short As[TM * ASTRIDE];
    __shared__ __align__(16) unsigned short Ws[NB_ * BSTRIDE];
    __shared__ float Ct[TM][NB_ + 4];

    const int tid  = threadIdx.x;
    const int lane = tid & 31;
    const int wave = tid >> 5;
    const int wm   = wave & 3;
    const int wn   = wave >> 2;
    const int block_m = blockIdx.x * TM;
    const int l  = lane & 15;
    const int kh = lane >> 4;

    floatx8 acc[2] = {};

    for (int k0 = 0; k0 < H_; k0 += TK) {
        // A tile: 64 x 64 (2 x 16B/thread);  W tile: 64 x 64 (2 x 16B/thread)
#pragma unroll
        for (int it = 0; it < 2; ++it) {
            int idx = tid + it * 256;
            int r = idx >> 3;
            int c = (idx & 7) * 8;
            copy16(A + (size_t)(block_m + r) * H_ + k0 + c, &As[r * ASTRIDE + c]);
            copy16(W + (size_t)r * H_ + k0 + c,             &Ws[r * BSTRIDE + c]);
        }
        wait_async();
        __syncthreads();

#pragma unroll
        for (int ks = 0; ks < TK; ks += 32) {
            FragU a;
            const unsigned short* pa = &As[(wm * 16 + l) * ASTRIDE + ks];
            a.u[0] = *(const uint4*)(pa + kh * 16);
            a.u[1] = *(const uint4*)(pa + kh * 16 + 8);
#pragma unroll
            for (int j = 0; j < 2; ++j) {
                FragU bm;
                const unsigned short* pb = &Ws[(wn * 32 + j * 16 + l) * BSTRIDE + ks + kh * 16];
                bm.u[0] = *(const uint4*)(pb);
                bm.u[1] = *(const uint4*)(pb + 8);
                acc[j] = __builtin_amdgcn_wmma_f32_16x16x32_bf16(
                    false, a.v, false, bm.v, (short)0, acc[j], false, false);
            }
        }
        __syncthreads();
    }

    // epilogue: bias add, write logits (strided into [B,T,NB]) + stage for argmax
    const int rl0 = wm * 16 + kh * 8;
#pragma unroll
    for (int j = 0; j < 2; ++j) {
        int col = wn * 32 + j * 16 + l;
        float bv = bias[col];
#pragma unroll
        for (int r = 0; r < 8; ++r) {
            float v  = acc[j][r] + bv;
            int   rl = rl0 + r;
            out[(size_t)(block_m + rl) * T_ * NB_ + (size_t)t * NB_ + col] = v;
            Ct[rl][col] = v;
        }
    }
    __syncthreads();

    if (tid < TM) {
        const float* rowp = Ct[tid];
        float bv = rowp[0];
        int   bi = 0;
#pragma unroll
        for (int j = 1; j < NB_; ++j) {
            float v = rowp[j];
            if (v > bv) { bv = v; bi = j; }
        }
        prev[block_m + tid] = bi;
    }
}

// ---------------------------------------------------------------------------
// Fused GRU update + LayerNorm + output-concat.  One block per batch row.
// ---------------------------------------------------------------------------
__global__ __launch_bounds__(256)
void gru_ln_kernel(const float* __restrict__ GX, const float* __restrict__ GH,
                   float* __restrict__ Hf, unsigned short* __restrict__ Hbf,
                   unsigned short* __restrict__ Ao,
                   const float* __restrict__ ctx, int t,
                   const float* __restrict__ ln_g, const float* __restrict__ ln_b) {
    const int b   = blockIdx.x;
    const int tid = threadIdx.x;
    __shared__ float red1[256];
    __shared__ float red2[256];

    const size_t base3 = (size_t)b * 3 * H_;
    const size_t baseH = (size_t)b * H_;

    float hn_loc[4];
    float s = 0.f, s2 = 0.f;
#pragma unroll
    for (int e = 0; e < 4; ++e) {
        int i  = tid + e * 256;
        float xr = GX[base3 + i];
        float xz = GX[base3 + H_ + i];
        float xn = GX[base3 + 2 * H_ + i];
        float hr = GH[base3 + i];
        float hz = GH[base3 + H_ + i];
        float hg = GH[base3 + 2 * H_ + i];
        float r  = sigmoidf(xr + hr);
        float z  = sigmoidf(xz + hz);
        float n  = tanhf(xn + r * hg);
        float h  = Hf[baseH + i];
        float hv = (1.0f - z) * n + z * h;
        Hf[baseH + i]  = hv;
        Hbf[baseH + i] = f2bf(hv);
        hn_loc[e] = hv;
        s  += hv;
        s2 += hv * hv;
    }
    red1[tid] = s;
    red2[tid] = s2;
    __syncthreads();
    for (int off = 128; off > 0; off >>= 1) {
        if (tid < off) {
            red1[tid] += red1[tid + off];
            red2[tid] += red2[tid + off];
        }
        __syncthreads();
    }
    float mean = red1[0] * (1.0f / H_);
    float var  = red2[0] * (1.0f / H_) - mean * mean;
    float inv  = rsqrtf(var + LN_EPS);

    const size_t baseA = (size_t)b * (H_ + D_);
#pragma unroll
    for (int e = 0; e < 4; ++e) {
        int i = tid + e * 256;
        float v = (hn_loc[e] - mean) * inv * ln_g[i] + ln_b[i];
        Ao[baseA + i] = f2bf(v);
    }
#pragma unroll
    for (int e = 0; e < 2; ++e) {
        int d = tid + e * 256;
        Ao[baseA + H_ + d] = f2bf(ctx[(size_t)b * T_ * D_ + (size_t)t * D_ + d]);
    }
}

// ---------------------------------------------------------------------------
// Host-side orchestration
// ---------------------------------------------------------------------------
extern "C" void kernel_launch(void* const* d_in, const int* in_sizes, int n_in,
                              void* d_out, int out_size, void* d_ws, size_t ws_size,
                              hipStream_t stream) {
    const float* ctx     = (const float*)d_in[0];
    const int*   hist    = (const int*)  d_in[1];
    const float* emb     = (const float*)d_in[2];
    const float* W_in    = (const float*)d_in[3];
    const float* b_in    = (const float*)d_in[4];
    const float* W_init  = (const float*)d_in[5];
    const float* b_init  = (const float*)d_in[6];
    const float* W_ih    = (const float*)d_in[7];
    const float* b_ih    = (const float*)d_in[8];
    const float* W_hh    = (const float*)d_in[9];
    const float* b_hh    = (const float*)d_in[10];
    const float* ln_g    = (const float*)d_in[11];
    const float* ln_b    = (const float*)d_in[12];
    const float* W_o1    = (const float*)d_in[13];
    const float* b_o1    = (const float*)d_in[14];
    const float* W_o2    = (const float*)d_in[15];
    const float* b_o2    = (const float*)d_in[16];
    float* out = (float*)d_out;

    // --- workspace carve-up (bf16 stored as ushort) ---
    char* ws = (char*)d_ws;
    auto carve = [&](size_t bytes) -> char* {
        char* p = ws;
        ws += (bytes + 255) & ~(size_t)255;
        return p;
    };
    unsigned short* Wbf_in   = (unsigned short*)carve((size_t)H_ * 2 * D_ * 2);
    unsigned short* Wbf_init = (unsigned short*)carve((size_t)H_ * 2 * D_ * 2);
    unsigned short* Wbf_ih   = (unsigned short*)carve((size_t)3 * H_ * H_ * 2);
    unsigned short* Wbf_hh   = (unsigned short*)carve((size_t)3 * H_ * H_ * 2);
    unsigned short* Wbf_o1   = (unsigned short*)carve((size_t)H_ * (H_ + D_) * 2);
    unsigned short* Wbf_o2   = (unsigned short*)carve((size_t)NB_ * H_ * 2);
    unsigned short* Ain      = (unsigned short*)carve((size_t)B_ * 2 * D_ * 2);
    unsigned short* Xbf      = (unsigned short*)carve((size_t)B_ * H_ * 2);
    unsigned short* Hbf      = (unsigned short*)carve((size_t)B_ * H_ * 2);
    unsigned short* Ao       = (unsigned short*)carve((size_t)B_ * (H_ + D_) * 2);
    unsigned short* Obf      = (unsigned short*)carve((size_t)B_ * H_ * 2);
    float*          Hf       = (float*)carve((size_t)B_ * H_ * 4);
    float*          GXf      = (float*)carve((size_t)B_ * 3 * H_ * 4);
    float*          GHf      = (float*)carve((size_t)B_ * 3 * H_ * 4);
    int*            prev     = (int*)carve((size_t)B_ * 4);
    (void)ws_size; (void)in_sizes; (void)n_in; (void)out_size;

    auto cvt = [&](const float* s, unsigned short* d, int n) {
        cvt_bf16_kernel<<<(n + 255) / 256, 256, 0, stream>>>(s, d, n);
    };
    cvt(W_in,   Wbf_in,   H_ * 2 * D_);
    cvt(W_init, Wbf_init, H_ * 2 * D_);
    cvt(W_ih,   Wbf_ih,   3 * H_ * H_);
    cvt(W_hh,   Wbf_hh,   3 * H_ * H_);
    cvt(W_o1,   Wbf_o1,   H_ * (H_ + D_));
    cvt(W_o2,   Wbf_o2,   NB_ * H_);

    auto gemm = [&](const unsigned short* A, int lda,
                    const unsigned short* Wt, int ldw, const float* bias,
                    float* Cf, long long ldc,
                    unsigned short* Cbf, int ldcbf,
                    int N, int K, int act) {
        dim3 grid(N / TN, B_ / TM);
        gemm_bf16_wmma<<<grid, dim3(256), 0, stream>>>(
            A, lda, Wt, ldw, bias, Cf, ldc, Cbf, ldcbf, N, K, act);
    };

    // h0 = tanh([ctx_mean, hist_emb] @ W_init^T + b_init)
    init_kernel<<<(B_ * D_) / 256, 256, 0, stream>>>(ctx, hist, emb, Ain, prev);
    gemm(Ain, 2 * D_, Wbf_init, 2 * D_, b_init, Hf, H_, Hbf, H_, H_, 2 * D_, /*tanh*/2);

    for (int t = 0; t < T_; ++t) {
        // x = relu([ctx_t, emb[prev]] @ W_in^T + b_in)
        concat_in_kernel<<<(B_ * 2 * D_) / 256, 256, 0, stream>>>(ctx, emb, prev, Ain, t);
        gemm(Ain, 2 * D_, Wbf_in, 2 * D_, b_in, nullptr, 0, Xbf, H_, H_, 2 * D_, /*relu*/1);
        // gx = x @ W_ih^T + b_ih ; gh = h @ W_hh^T + b_hh
        gemm(Xbf, H_, Wbf_ih, H_, b_ih, GXf, 3 * H_, nullptr, 0, 3 * H_, H_, 0);
        gemm(Hbf, H_, Wbf_hh, H_, b_hh, GHf, 3 * H_, nullptr, 0, 3 * H_, H_, 0);
        // GRU gates + layernorm + [h_norm, ctx_t] pack
        gru_ln_kernel<<<B_, 256, 0, stream>>>(GXf, GHf, Hf, Hbf, Ao, ctx, t, ln_g, ln_b);
        // o = relu([h_norm, ctx_t] @ W_o1^T + b_o1)
        gemm(Ao, H_ + D_, Wbf_o1, H_ + D_, b_o1, nullptr, 0, Obf, H_, H_, H_ + D_, 1);
        // logits_t = o @ W_o2^T + b_o2, fused with argmax -> prev
        logits_argmax_kernel<<<B_ / TM, 256, 0, stream>>>(Obf, Wbf_o2, b_o2, out, t, prev);
    }
}
